// MultiHeadedAttention_8143257993712
// MI455X (gfx1250) — matrix operand order repaired and checked by reading, not compiled, not run on hardware
//
#include <hip/hip_runtime.h>

// ---------------------------------------------------------------------------
// MultiHeadedAttention for MI455X (gfx1250), f16 WMMA with f32 accumulation.
// B=2, S=2048, H=16, D=64, DM=1024.
// K-tiles staged with GLOBAL_LOAD_ASYNC_TO_LDS_B128 (ASYNCcnt) into
// double-buffered LDS; fragments read as ds_load_b128.
// ---------------------------------------------------------------------------

typedef _Float16 v16h __attribute__((ext_vector_type(16)));
typedef _Float16 v8h  __attribute__((ext_vector_type(8)));
typedef _Float16 v4h  __attribute__((ext_vector_type(4)));
typedef float    v8f  __attribute__((ext_vector_type(8)));

#define WMMA16(a, b, c) \
  __builtin_amdgcn_wmma_f32_16x16x32_f16(false, (a), false, (b), (short)0, (c), false, false)

// Async 16-byte global->LDS copy (per-lane addresses), ISA §15.18 opcode 98.
__device__ inline void async_ld16(void* lds_dst, const void* gsrc) {
  asm volatile("global_load_async_to_lds_b128 %0, %1, off"
               :
               : "v"((unsigned)(unsigned long long)lds_dst),
                 "v"((unsigned long long)gsrc)
               : "memory");
}
__device__ inline void wait_async0() {
  asm volatile("s_wait_asynccnt 0x0" ::: "memory");
}

__device__ inline v16h cat8(v8h lo, v8h hi) {
  return __builtin_shufflevector(lo, hi, 0, 1, 2, 3, 4, 5, 6, 7, 8, 9, 10, 11,
                                 12, 13, 14, 15);
}

// ----- fragment loaders (ISA 7.12.2 layouts, wave32) -----------------------
__device__ inline v16h load_frag_a(const _Float16* base, int pitch) {
  const int lane = threadIdx.x & 31;
  const _Float16* p = base + (lane & 15) * pitch + ((lane < 16) ? 0 : 8);
  return cat8(*(const v8h*)p, *(const v8h*)(p + 16));
}
// B fragment from src[n][k] row-major: per-lane k = 8h..8h+7, 16+8h..16+8h+7.
__device__ inline v16h load_frag_bt(const _Float16* base, int pitch) {
  const int lane = threadIdx.x & 31;
  const _Float16* p = base + (lane & 15) * pitch + ((lane < 16) ? 0 : 8);
  return cat8(*(const v8h*)p, *(const v8h*)(p + 16));
}

// ----- f32 -> f16 elementwise conversion -----------------------------------
__global__ __launch_bounds__(256) void f32_to_f16_kernel(
    const float* __restrict__ x, _Float16* __restrict__ y, int n4) {
  int i = blockIdx.x * 256 + threadIdx.x;
  if (i >= n4) return;
  float4 f = ((const float4*)x)[i];
  v4h o;
  o[0] = (_Float16)f.x; o[1] = (_Float16)f.y;
  o[2] = (_Float16)f.z; o[3] = (_Float16)f.w;
  ((v4h*)y)[i] = o;
}

// ----- GEMM: Y[4096][1024] = X f16 @ Wh f16 [1024][1024]^T + bias ----------
// Block tile 64(M) x 128(N), 8 waves, wave = 32x32 (2x2 WMMA), K-step 32.
// Double-buffered LDS, filled by async b128 copies.
// MODE 0: f16 head-split [B,H,S,D]    MODE 1: f32 row-major.

template <int MODE>
__global__ __launch_bounds__(256) void gemm_wmma_kernel(
    const _Float16* __restrict__ X, const _Float16* __restrict__ Wh,
    const float* __restrict__ bias, void* __restrict__ outp) {
  __shared__ _Float16 As[2][64 * 40];   // [64 m][32 k] pitch 40
  __shared__ _Float16 Bs[2][128 * 40];  // [128 n][32 k] pitch 40

  const int tid  = threadIdx.x;
  const int lane = tid & 31;
  const int wv   = tid >> 5;
  const int wm   = wv >> 2;
  const int wn   = wv & 3;
  const int m0   = blockIdx.y * 64;
  const int n0   = blockIdx.x * 128;
  const int hi8  = (lane < 16) ? 0 : 8;
  const int nl   = lane & 15;

  v8f zero = {};
  v8f acc[2][2];
  acc[0][0] = zero; acc[0][1] = zero; acc[1][0] = zero; acc[1][1] = zero;

  const int ar = tid >> 2, aseg = tid & 3;  // A: 64 rows x 4 16B segs
  const _Float16* aS = X + (size_t)(m0 + ar) * 1024 + aseg * 8;

  auto issue = [&](int k0, int buf) {
    async_ld16(&As[buf][ar * 40 + aseg * 8], aS + k0);
#pragma unroll
    for (int u = 0; u < 2; ++u) {  // B: 128 rows x 4 16B segs
      const int idx = tid * 2 + u;
      const int n = idx >> 2, sg = idx & 3;
      async_ld16(&Bs[buf][n * 40 + sg * 8],
                 Wh + (size_t)(n0 + n) * 1024 + k0 + sg * 8);
    }
  };

  issue(0, 0);
  wait_async0();
  __syncthreads();

  for (int kc = 0; kc < 32; ++kc) {
    const int cur = kc & 1;
    const bool more = kc < 31;
    if (more) issue((kc + 1) * 32, cur ^ 1);  // overlap copy with math

    v16h a0 = load_frag_a(&As[cur][(wm * 32 + 0) * 40], 40);
    v16h a1 = load_frag_a(&As[cur][(wm * 32 + 16) * 40], 40);
    v16h b0 = load_frag_bt(&Bs[cur][(wn * 32 + 0) * 40], 40);
    v16h b1 = load_frag_bt(&Bs[cur][(wn * 32 + 16) * 40], 40);
    acc[0][0] = WMMA16(a0, b0, acc[0][0]);
    acc[0][1] = WMMA16(a0, b1, acc[0][1]);
    acc[1][0] = WMMA16(a1, b0, acc[1][0]);
    acc[1][1] = WMMA16(a1, b1, acc[1][1]);

    if (more) {
      wait_async0();
      __syncthreads();
    }
  }

#pragma unroll
  for (int i = 0; i < 2; ++i)
#pragma unroll
    for (int j = 0; j < 2; ++j)
#pragma unroll
      for (int e = 0; e < 8; ++e) {
        const int m = m0 + wm * 32 + i * 16 + e + hi8;
        const int n = n0 + wn * 32 + j * 16 + nl;
        const float val = acc[i][j][e] + bias[n];
        if constexpr (MODE == 0) {
          const int b = m >> 11, s = m & 2047, h = n >> 6, d = n & 63;
          ((_Float16*)outp)[(((size_t)(b * 16 + h) * 2048 + s) << 6) + d] =
              (_Float16)val;
        } else {
          ((float*)outp)[(size_t)m * 1024 + n] = val;
        }
      }
}

// ----- causal attention, faithful to the reference's masked softmax --------
// grid: (S/64 query tiles, B*H). Block 256 = 8 waves.
// Q and K tiles staged by async copies (K double-buffered); V is transposed
// into LDS via registers (async copies cannot transpose).

__global__ __launch_bounds__(256) void attn_kernel(
    const _Float16* __restrict__ q, const _Float16* __restrict__ k,
    const _Float16* __restrict__ v, _Float16* __restrict__ attn) {
  __shared__ _Float16 Qt[64 * 72];      // [64 q][64 d]
  __shared__ _Float16 Kt[2][32 * 72];   // [32 key][64 d]
  __shared__ _Float16 Vtt[2][64 * 40];  // [64 d][32 key]
  __shared__ _Float16 Pt[64 * 40];      // [64 q][32 key]
  __shared__ float    rs[64];

  const int tid  = threadIdx.x;
  const int lane = tid & 31;
  const int wv   = tid >> 5;
  const int qs   = wv & 3;
  const int ks   = wv >> 2;
  const int dh   = wv >> 2;
  const int hi8  = (lane < 16) ? 0 : 8;
  const int nl   = lane & 15;

  const int qt = blockIdx.x;
  const int bh = blockIdx.y;
  const int s0 = qt * 64;
  const size_t head = (size_t)bh * 2048 * 64;

  if (tid < 64) rs[tid] = 0.0f;

  const int nkt = 2 * qt + 2;
  const int kr = tid >> 3, kseg = tid & 7;
  const _Float16* kSrc = k + head + (size_t)kr * 64 + kseg * 8;
  const _Float16* vSrc = v + head + (size_t)kr * 64 + kseg * 8;

  // Q tile (async), K tile 0 (async), V tile 0 (register transpose)
#pragma unroll
  for (int p = 0; p < 2; ++p) {
    const int idx = tid + p * 256;
    const int r = idx >> 3, seg = idx & 7;
    async_ld16(&Qt[r * 72 + seg * 8],
               q + head + (size_t)(s0 + r) * 64 + seg * 8);
  }
  async_ld16(&Kt[0][kr * 72 + kseg * 8], kSrc);
  {
    uint4 vR = *(const uint4*)vSrc;
    v8h vh = *(v8h*)&vR;
#pragma unroll
    for (int i = 0; i < 8; ++i) Vtt[0][(kseg * 8 + i) * 40 + kr] = vh[i];
  }
  wait_async0();
  __syncthreads();

  v8f zero = {};
  v8f oacc[2];
  oacc[0] = zero; oacc[1] = zero;

  for (int kt = 0; kt < nkt; ++kt) {
    const int cur = kt & 1;
    const bool more = kt + 1 < nkt;
    uint4 vN;
    if (more) {
      async_ld16(&Kt[cur ^ 1][kr * 72 + kseg * 8],
                 kSrc + (size_t)(kt + 1) * 2048);
      vN = *(const uint4*)(vSrc + (size_t)(kt + 1) * 2048);
    }

    // logits [16 q x 16 key], K-dim = d in two 32-wide steps
    v8f sacc = zero;
#pragma unroll
    for (int d0 = 0; d0 < 64; d0 += 32) {
      v16h a = load_frag_a(&Qt[(qs * 16) * 72 + d0], 72);
      v16h b = load_frag_bt(&Kt[cur][(ks * 16) * 72 + d0], 72);
      sacc = WMMA16(a, b, sacc);
    }
    // masked exp -> Pt (f16); masked lanes produce exactly 0 (exp(-1e4))
    {
      const int colk = ks * 16 + nl;
      const int key  = kt * 32 + colk;
#pragma unroll
      for (int e = 0; e < 8; ++e) {
        const int rloc = qs * 16 + e + hi8;
        const float pe =
            (key <= s0 + rloc) ? __expf(sacc[e] * 0.125f) : 0.0f;
        Pt[rloc * 40 + colk] = (_Float16)pe;
      }
    }
    __syncthreads();

    // denominator (single writer per row, vector reads)
    if (tid < 64) {
      const v8h* pr = (const v8h*)&Pt[tid * 40];
      float a = 0.0f;
#pragma unroll
      for (int c = 0; c < 4; ++c) {
        v8h x = pr[c];
#pragma unroll
        for (int i = 0; i < 8; ++i) a += (float)x[i];
      }
      rs[tid] += a;
    }

    // O += P @ V
    {
      v16h a2 = load_frag_a(&Pt[(qs * 16) * 40], 40);
#pragma unroll
      for (int t = 0; t < 2; ++t) {
        v16h b2 = load_frag_bt(&Vtt[cur][(dh * 32 + t * 16) * 40], 40);
        oacc[t] = WMMA16(a2, b2, oacc[t]);
      }
    }

    if (more) {
      v8h vh = *(v8h*)&vN;
#pragma unroll
      for (int i = 0; i < 8; ++i)
        Vtt[cur ^ 1][(kseg * 8 + i) * 40 + kr] = vh[i];
      wait_async0();
      __syncthreads();
    }
  }

  // divide by denominator, store merged-heads f16 [B, S, H*D]
  const int bB = bh >> 4, hh = bh & 15;
#pragma unroll
  for (int t = 0; t < 2; ++t)
#pragma unroll
    for (int e = 0; e < 8; ++e) {
      const int rloc  = qs * 16 + e + hi8;
      const float den = rs[rloc];
      const int dcol  = dh * 32 + t * 16 + nl;
      attn[((size_t)(bB * 2048 + s0 + rloc)) * 1024 + hh * 64 + dcol] =
          (_Float16)(oacc[t][e] / den);
    }
}

// ---------------------------------------------------------------------------

extern "C" void kernel_launch(void* const* d_in, const int* in_sizes, int n_in,
                              void* d_out, int out_size, void* d_ws,
                              size_t ws_size, hipStream_t stream) {
  const float* hs = (const float*)d_in[0];
  const float* Wq = (const float*)d_in[1];
  const float* bq = (const float*)d_in[2];
  const float* Wk = (const float*)d_in[3];
  const float* bk = (const float*)d_in[4];
  const float* Wv = (const float*)d_in[5];
  const float* bv = (const float*)d_in[6];
  const float* Wy = (const float*)d_in[7];
  const float* by = (const float*)d_in[8];

  char* ws = (char*)d_ws;
  const size_t SZ = (size_t)4096 * 1024 * sizeof(_Float16);  // 8 MB
  const size_t WZ = (size_t)1024 * 1024 * sizeof(_Float16);  // 2 MB
  _Float16* x16  = (_Float16*)(ws + 0 * SZ);
  _Float16* qbuf = (_Float16*)(ws + 1 * SZ);  // [B,H,S,D]
  _Float16* kbuf = (_Float16*)(ws + 2 * SZ);
  _Float16* vbuf = (_Float16*)(ws + 3 * SZ);
  _Float16* abuf = (_Float16*)(ws + 4 * SZ);  // attended [B,S,DM]
  _Float16* wqh  = (_Float16*)(ws + 5 * SZ + 0 * WZ);
  _Float16* wkh  = (_Float16*)(ws + 5 * SZ + 1 * WZ);
  _Float16* wvh  = (_Float16*)(ws + 5 * SZ + 2 * WZ);
  _Float16* wyh  = (_Float16*)(ws + 5 * SZ + 3 * WZ);

  f32_to_f16_kernel<<<4096, 256, 0, stream>>>(hs, x16, 4096 * 1024 / 4);
  f32_to_f16_kernel<<<1024, 256, 0, stream>>>(Wq, wqh, 1024 * 1024 / 4);
  f32_to_f16_kernel<<<1024, 256, 0, stream>>>(Wk, wkh, 1024 * 1024 / 4);
  f32_to_f16_kernel<<<1024, 256, 0, stream>>>(Wv, wvh, 1024 * 1024 / 4);
  f32_to_f16_kernel<<<1024, 256, 0, stream>>>(Wy, wyh, 1024 * 1024 / 4);

  dim3 ggrid(1024 / 128, 4096 / 64);
  gemm_wmma_kernel<0><<<ggrid, 256, 0, stream>>>(x16, wqh, bq, qbuf);
  gemm_wmma_kernel<0><<<ggrid, 256, 0, stream>>>(x16, wkh, bk, kbuf);
  gemm_wmma_kernel<0><<<ggrid, 256, 0, stream>>>(x16, wvh, bv, vbuf);

  attn_kernel<<<dim3(2048 / 64, 2 * 16), 256, 0, stream>>>(qbuf, kbuf, vbuf,
                                                           abuf);

  gemm_wmma_kernel<1><<<ggrid, 256, 0, stream>>>(abuf, wyh, by, d_out);
}
